// LocalDecoder_3659312136619
// MI455X (gfx1250) — compile-verified, matrix operand order checked by reading, not run on hardware
//
#include <hip/hip_runtime.h>
#include <hip/hip_bf16.h>

// ---------------------------------------------------------------------------
// Problem constants (from reference)
// ---------------------------------------------------------------------------
#define BB      4
#define LL      4096
#define PP      1024
#define DM      1024          // D_MODEL
#define DG      2048          // D_GLOBAL
#define EPS_LP  1.0e-3f       // EPSILON
#define RMS_EPS 1.0e-5f

typedef __attribute__((ext_vector_type(16))) _Float16 v16h;
typedef __attribute__((ext_vector_type(8)))  float    v8f;
typedef __attribute__((ext_vector_type(4)))  _Float16 hh4;

// ---------------------------------------------------------------------------
// Kernel 1: rmsnorm(embeds, g_res) -> f16 A-matrix  [16384 x 1024]
// ---------------------------------------------------------------------------
__global__ void __launch_bounds__(256)
k_rmsnorm_a(const float* __restrict__ x, const float* __restrict__ g,
            _Float16* __restrict__ y) {
    const int row  = blockIdx.x;
    const int t    = threadIdx.x;
    const int lane = t & 31, wid = t >> 5;

    const float4 v  = ((const float4*)(x + (size_t)row * DM))[t];
    float ss = v.x * v.x + v.y * v.y + v.z * v.z + v.w * v.w;
    #pragma unroll
    for (int o = 16; o > 0; o >>= 1) ss += __shfl_down(ss, o, 32);

    __shared__ float sred[8];
    __shared__ float stot;
    if (lane == 0) sred[wid] = ss;
    __syncthreads();
    if (t == 0) {
        float s = 0.f;
        #pragma unroll
        for (int i = 0; i < 8; ++i) s += sred[i];
        stot = s;
    }
    __syncthreads();

    const float scale = rsqrtf(stot * (1.0f / DM) + RMS_EPS);
    const float4 gv = ((const float4*)g)[t];
    hh4 o;
    o[0] = (_Float16)(v.x * scale * gv.x);
    o[1] = (_Float16)(v.y * scale * gv.y);
    o[2] = (_Float16)(v.z * scale * gv.z);
    o[3] = (_Float16)(v.w * scale * gv.w);
    ((hh4*)(y + (size_t)row * DM))[t] = o;
}

// ---------------------------------------------------------------------------
// Kernel 2: W_in (f32, [1024 x 1024] row-major) -> f16
// ---------------------------------------------------------------------------
__global__ void __launch_bounds__(256)
k_cvt_w(const float* __restrict__ w, _Float16* __restrict__ wh) {
    const size_t i = (size_t)blockIdx.x * 1024 + (size_t)threadIdx.x * 4;
    const float4 v = *(const float4*)(w + i);
    hh4 o;
    o[0] = (_Float16)v.x; o[1] = (_Float16)v.y;
    o[2] = (_Float16)v.z; o[3] = (_Float16)v.w;
    *(hh4*)(wh + i) = o;
}

// ---------------------------------------------------------------------------
// Kernel 3: rmsnorm(patch_embeds, g_init) over D_GLOBAL=2048,
// keep first 1024 dims as f32 -> hpn [4 x 1024 x 1024]
// ---------------------------------------------------------------------------
__global__ void __launch_bounds__(256)
k_rmsnorm_p(const float* __restrict__ x, const float* __restrict__ g,
            float* __restrict__ y) {
    const int row  = blockIdx.x;           // b*P + p
    const int t    = threadIdx.x;
    const int lane = t & 31, wid = t >> 5;

    const float4* xr = (const float4*)(x + (size_t)row * DG);
    const float4 v0 = xr[t];
    const float4 v1 = xr[t + 256];
    float ss = v0.x * v0.x + v0.y * v0.y + v0.z * v0.z + v0.w * v0.w
             + v1.x * v1.x + v1.y * v1.y + v1.z * v1.z + v1.w * v1.w;
    #pragma unroll
    for (int o = 16; o > 0; o >>= 1) ss += __shfl_down(ss, o, 32);

    __shared__ float sred[8];
    __shared__ float stot;
    if (lane == 0) sred[wid] = ss;
    __syncthreads();
    if (t == 0) {
        float s = 0.f;
        #pragma unroll
        for (int i = 0; i < 8; ++i) s += sred[i];
        stot = s;
    }
    __syncthreads();

    const float scale = rsqrtf(stot * (1.0f / DG) + RMS_EPS);
    const float4 gv = ((const float4*)g)[t];   // dims 4t..4t+3 (< 1024)
    float4 o;
    o.x = v0.x * scale * gv.x; o.y = v0.y * scale * gv.y;
    o.z = v0.z * scale * gv.z; o.w = v0.w * scale * gv.w;
    ((float4*)(y + (size_t)row * DM))[t] = o;
}

// ---------------------------------------------------------------------------
// Kernel 4: WMMA GEMM  out[m,n] = sum_k A[m,k]*W[n,k] + bias[n]   (fp32 out)
// A: f16 [16384 x 1024] row-major, W: f16 [1024 x 1024] row-major.
// Block = 256 thr = 8 waves arranged 4(M) x 2(N); block tile 128 x 128.
// Wave tile 32(M) x 64(N): 2 A frags x 4 B frags -> 8 WMMAs / K-step.
// ---------------------------------------------------------------------------
__device__ __forceinline__ v16h ld_frag(const _Float16* __restrict__ p) {
    union { v16h v; uint4 q[2]; } f;
    f.q[0] = *(const uint4*)(p);        // 8 halves: K = c0 .. c0+7
    f.q[1] = *(const uint4*)(p + 16);   // 8 halves: K = c0+16 .. c0+23
    return f.v;
}

__global__ void __launch_bounds__(256)
k_gemm(const _Float16* __restrict__ A, const _Float16* __restrict__ Bw,
       const float* __restrict__ bias, float* __restrict__ out) {
    const int wave = threadIdx.x >> 5;
    const int lane = threadIdx.x & 31;
    const int wm = wave & 3;            // 4 waves along M
    const int wn = wave >> 2;           // 2 waves along N
    const int m0 = blockIdx.y * 128 + wm * 32;
    const int n0 = blockIdx.x * 128 + wn * 64;

    const int r  = lane & 15;           // row (A) / col (B) within tile
    const int c0 = (lane >> 4) << 3;    // K sub-offset {0,8}

    const _Float16* ap0 = A  + (size_t)(m0 + r) * DM + c0;
    const _Float16* ap1 = ap0 + (size_t)16 * DM;
    const _Float16* bp0 = Bw + (size_t)(n0 + r) * DM + c0;
    const _Float16* bp1 = bp0 + (size_t)16 * DM;
    const _Float16* bp2 = bp0 + (size_t)32 * DM;
    const _Float16* bp3 = bp0 + (size_t)48 * DM;

    v8f c00 = {}, c01 = {}, c02 = {}, c03 = {};
    v8f c10 = {}, c11 = {}, c12 = {}, c13 = {};

    #pragma unroll 2
    for (int k = 0; k < DM; k += 32) {
        // unconditional speculative prefetch, all cache levels (WGP scope);
        // addresses stay inside the workspace (A is followed by W in ws)
        __builtin_prefetch(ap0 + k + 128, 0, 3);
        __builtin_prefetch(bp0 + k + 128, 0, 3);

        const v16h a0 = ld_frag(ap0 + k);
        const v16h a1 = ld_frag(ap1 + k);
        const v16h b0 = ld_frag(bp0 + k);
        const v16h b1 = ld_frag(bp1 + k);
        const v16h b2 = ld_frag(bp2 + k);
        const v16h b3 = ld_frag(bp3 + k);

        c00 = __builtin_amdgcn_wmma_f32_16x16x32_f16(false, a0, false, b0,
                                                     (short)0, c00, false, false);
        c01 = __builtin_amdgcn_wmma_f32_16x16x32_f16(false, a0, false, b1,
                                                     (short)0, c01, false, false);
        c02 = __builtin_amdgcn_wmma_f32_16x16x32_f16(false, a0, false, b2,
                                                     (short)0, c02, false, false);
        c03 = __builtin_amdgcn_wmma_f32_16x16x32_f16(false, a0, false, b3,
                                                     (short)0, c03, false, false);
        c10 = __builtin_amdgcn_wmma_f32_16x16x32_f16(false, a1, false, b0,
                                                     (short)0, c10, false, false);
        c11 = __builtin_amdgcn_wmma_f32_16x16x32_f16(false, a1, false, b1,
                                                     (short)0, c11, false, false);
        c12 = __builtin_amdgcn_wmma_f32_16x16x32_f16(false, a1, false, b2,
                                                     (short)0, c12, false, false);
        c13 = __builtin_amdgcn_wmma_f32_16x16x32_f16(false, a1, false, b3,
                                                     (short)0, c13, false, false);
    }

    // C/D layout: vgpr i -> row i + (lane>=16 ? 8 : 0), col = lane&15
    const int rowoff = (lane >> 4) << 3;
    const int col    = lane & 15;
    const float bv0 = bias[n0      + col];
    const float bv1 = bias[n0 + 16 + col];
    const float bv2 = bias[n0 + 32 + col];
    const float bv3 = bias[n0 + 48 + col];
    #pragma unroll
    for (int i = 0; i < 8; ++i) {
        float* o0 = out + (size_t)(m0 + rowoff + i) * DM + n0 + col;
        float* o1 = o0 + (size_t)16 * DM;
        o0[ 0] = c00[i] + bv0;
        o0[16] = c01[i] + bv1;
        o0[32] = c02[i] + bv2;
        o0[48] = c03[i] + bv3;
        o1[ 0] = c10[i] + bv0;
        o1[16] = c11[i] + bv1;
        o1[32] = c12[i] + bv2;
        o1[48] = c13[i] + bv3;
    }
}

// ---------------------------------------------------------------------------
// Kernel 5: depool scan.  s[p] = a*s[p-1] + u[p],  u = (dt*bcoef)*(hpn/dt).
// One thread per (b, d): 4096 independent recurrences of length 1024.
// ---------------------------------------------------------------------------
__global__ void __launch_bounds__(256)
k_scan(const float* __restrict__ hpn, float* __restrict__ dp) {
    const int d = blockIdx.x * 256 + threadIdx.x;   // 0..1023
    const int b = blockIdx.y;

    const float em     = -expm1f(-EPS_LP);   // -expm1(logp)
    const float dt     = -logf(em);
    const float bcoef  = expf(-EPS_LP);
    const float a      = expf(-dt);
    const float uc     = dt * bcoef;
    const float inv_dt = 1.0f / dt;

    const float* col = hpn + ((size_t)b << 20) + d;
    float*       o   = dp  + ((size_t)b << 20) + d;
    float s = 0.0f;
    for (int p = 0; p < PP; ++p) {
        const float u = uc * (col[(size_t)p << 10] * inv_dt);
        s = fmaf(a, s, u);
        o[(size_t)p << 10] = s;
    }
}

// ---------------------------------------------------------------------------
// Kernel 6a: bidx[b][p] = cumsum(patch_lens[b]) - 1
// ---------------------------------------------------------------------------
__global__ void k_bidx(const int* __restrict__ pl, int* __restrict__ bx) {
    const int b = threadIdx.x;
    if (b < BB) {
        int acc = 0;
        for (int p = 0; p < PP; ++p) {
            acc += pl[b * PP + p];
            bx[b * PP + p] = acc - 1;
        }
    }
}

// ---------------------------------------------------------------------------
// Kernel 6b: out[b,l,:] += depool[b, plug(b,l), :]
// plug = clamp(#{bidx<=l} - 1, 0, P-1) via binary search (bidx increasing)
// ---------------------------------------------------------------------------
__global__ void __launch_bounds__(256)
k_gather(const float* __restrict__ dp, const int* __restrict__ bx,
         float* __restrict__ out) {
    const int rowid = blockIdx.x;          // b*L + l
    const int b = rowid >> 12;
    const int l = rowid & (LL - 1);

    const int* bb = bx + b * PP;
    int lo = 0, hi = PP;
    while (lo < hi) {
        const int mid = (lo + hi) >> 1;
        if (bb[mid] <= l) lo = mid + 1; else hi = mid;
    }
    int plug = lo - 1;
    plug = plug < 0 ? 0 : (plug > PP - 1 ? PP - 1 : plug);

    const float4 add = ((const float4*)(dp + ((size_t)b << 20) +
                                        ((size_t)plug << 10)))[threadIdx.x];
    float4* o = (float4*)(out + (size_t)rowid * DM);
    float4 cur = o[threadIdx.x];
    cur.x += add.x; cur.y += add.y; cur.z += add.z; cur.w += add.w;
    o[threadIdx.x] = cur;
}

// ---------------------------------------------------------------------------
// Launch
// ---------------------------------------------------------------------------
extern "C" void kernel_launch(void* const* d_in, const int* in_sizes, int n_in,
                              void* d_out, int out_size, void* d_ws, size_t ws_size,
                              hipStream_t stream) {
    const float* embeds = (const float*)d_in[0];   // 4*4096*1024
    const float* patch  = (const float*)d_in[1];   // 4*1024*2048
    const int*   plens  = (const int*)d_in[2];     // 4*1024
    const float* W      = (const float*)d_in[3];   // 1024*1024
    const float* bias   = (const float*)d_in[4];   // 1024
    const float* gres   = (const float*)d_in[5];   // 1024
    const float* ginit  = (const float*)d_in[6];   // 2048
    float* out = (float*)d_out;                    // 4*4096*1024

    char* ws = (char*)d_ws;
    _Float16* Ah  = (_Float16*)(ws);               // 33,554,432 B
    _Float16* Wh  = (_Float16*)(ws + 33554432);    //  2,097,152 B
    float*    hpn = (float*)   (ws + 35651584);    // 16,777,216 B
    float*    dp  = (float*)   (ws + 52428800);    // 16,777,216 B
    int*      bx  = (int*)     (ws + 69206016);    //     16,384 B

    k_rmsnorm_a<<<BB * LL, 256, 0, stream>>>(embeds, gres, Ah);
    k_cvt_w    <<<1024,    256, 0, stream>>>(W, Wh);
    k_rmsnorm_p<<<BB * PP, 256, 0, stream>>>(patch, ginit, hpn);
    k_gemm     <<<dim3(DM / 128, (BB * LL) / 128), 256, 0, stream>>>(Ah, Wh, bias, out);
    k_scan     <<<dim3(DM / 256, BB), 256, 0, stream>>>(hpn, dp);
    k_bidx     <<<1, 32, 0, stream>>>(plens, bx);
    k_gather   <<<BB * LL, 256, 0, stream>>>(dp, bx, out);
}
